// Perceiver_28106265985541
// MI455X (gfx1250) — matrix operand order repaired and checked by reading, not compiled, not run on hardware
//
#include <hip/hip_runtime.h>
#include <hip/hip_bf16.h>
#include <math.h>

#define B_   4
#define SQ_  512
#define SKV_ 4096
#define D_   1024
#define H_   16
#define HD_  64
#define FFW_ 4096

typedef __attribute__((ext_vector_type(16))) __bf16 v16bf;
typedef __attribute__((ext_vector_type(8)))  float  v8f;

union BFrag { v16bf v; unsigned short u[16]; uint4 q[2]; };

__device__ __forceinline__ unsigned short f2bf(float f) {
  unsigned int x = __float_as_uint(f);
  unsigned int r = (x + 0x7fffu + ((x >> 16) & 1u)) >> 16;  // RNE
  return (unsigned short)r;
}

// elements [0..7] from p0, [8..15] from p1 (each 16B contiguous)
__device__ __forceinline__ v16bf load_frag2(const unsigned short* p0,
                                            const unsigned short* p1) {
  BFrag f;
  f.q[0] = *reinterpret_cast<const uint4*>(p0);
  f.q[1] = *reinterpret_cast<const uint4*>(p1);
  return f.v;
}

// A-fragment from f32 source: elems[0..7]=p[0..7], elems[8..15]=p[8..15]
__device__ __forceinline__ v16bf load_frag_f32(const float* p) {
  const float4* a = reinterpret_cast<const float4*>(p);
  float4 x0 = a[0], x1 = a[1], x2 = a[2], x3 = a[3];
  BFrag f;
  f.u[0]=f2bf(x0.x);  f.u[1]=f2bf(x0.y);  f.u[2]=f2bf(x0.z);  f.u[3]=f2bf(x0.w);
  f.u[4]=f2bf(x1.x);  f.u[5]=f2bf(x1.y);  f.u[6]=f2bf(x1.z);  f.u[7]=f2bf(x1.w);
  f.u[8]=f2bf(x2.x);  f.u[9]=f2bf(x2.y);  f.u[10]=f2bf(x2.z); f.u[11]=f2bf(x2.w);
  f.u[12]=f2bf(x3.x); f.u[13]=f2bf(x3.y); f.u[14]=f2bf(x3.z); f.u[15]=f2bf(x3.w);
  return f.v;
}

__device__ __forceinline__ v8f wmma_bf(v16bf a, v16bf b, v8f c) {
  return __builtin_amdgcn_wmma_f32_16x16x32_bf16(false, a, false, b, (short)0, c,
                                                 false, false);
}

// ---------------- LayerNorm: fp32 rows -> bf16 rows ----------------
__global__ void __launch_bounds__(256, 1)
ln_kernel(const float* __restrict__ x, const float* __restrict__ g,
          const float* __restrict__ b, unsigned short* __restrict__ y, int D) {
  int row = blockIdx.x;
  const float* xr = x + (size_t)row * D;
  __shared__ float red[32];
  int tid = threadIdx.x, lane = tid & 31, w = tid >> 5;
  float s = 0.f, s2 = 0.f;
  for (int i = tid; i < D; i += blockDim.x) { float v = xr[i]; s += v; s2 += v * v; }
  for (int o = 16; o > 0; o >>= 1) { s += __shfl_down(s, o, 32); s2 += __shfl_down(s2, o, 32); }
  if (lane == 0) { red[w] = s; red[8 + w] = s2; }
  __syncthreads();
  if (tid == 0) {
    float a = 0.f, c = 0.f;
    for (int i = 0; i < 8; ++i) { a += red[i]; c += red[8 + i]; }
    red[16] = a; red[17] = c;
  }
  __syncthreads();
  float mean = red[16] / (float)D;
  float var  = red[17] / (float)D - mean * mean;
  float inv  = rsqrtf(var + 1e-5f);
  unsigned short* yr = y + (size_t)row * D;
  for (int i = tid; i < D; i += blockDim.x)
    yr[i] = f2bf((xr[i] - mean) * inv * g[i] + b[i]);
}

// ---------------- Weight pack: f32 [K,N] -> bf16 B-fragment order ----------------
__global__ void __launch_bounds__(256, 1)
packw_kernel(const float* __restrict__ W, unsigned short* __restrict__ P, int N) {
  size_t t = (size_t)blockIdx.x * 256 + threadIdx.x;
  int j = (int)(t & 15);
  int lane = (int)((t >> 4) & 31);
  size_t g = t >> 9;
  int ntiles = N >> 4;
  int nt = (int)(g % ntiles);
  int kt = (int)(g / ntiles);
  int k = kt * 32 + ((lane & 16) ? 16 : 0) + j;
  int n = nt * 16 + (lane & 15);
  P[t] = f2bf(W[(size_t)k * N + n]);
}

// ------- bf16 WMMA GEMM, wave = 32x64 tile, register double-buffered K-loop -------
__global__ void __launch_bounds__(256, 1)
gemm_bf16_kernel(const unsigned short* __restrict__ A,
                 const unsigned short* __restrict__ Bp,
                 const float* __restrict__ bias,
                 const float* __restrict__ resid,
                 float* __restrict__ outf,
                 unsigned short* __restrict__ outb,
                 int N, int K, int act) {
  int lane = threadIdx.x & 31, w = threadIdx.x >> 5;
  int nChunks = N >> 9;                       // 512 cols per block
  int mt = blockIdx.x / nChunks;              // 32-row slab index
  int nc = blockIdx.x % nChunks;
  int n0 = (nc << 9) + (w << 6);              // this wave's 64-col slab
  int m = lane & 15;
  int kbA = (lane >> 4) << 4;                 // 0 or 16
  int ntiles = N >> 4;
  const unsigned short* Arow0 = A + (size_t)(mt * 32 + m) * K + kbA;
  const unsigned short* Arow1 = Arow0 + (size_t)16 * K;
  const size_t bstep = (size_t)ntiles * 512;  // elems per kt in packed B
  const unsigned short* Bbase = Bp + ((size_t)(n0 >> 4) * 32 + lane) * 16;

  v8f acc[8] = {v8f{}, v8f{}, v8f{}, v8f{}, v8f{}, v8f{}, v8f{}, v8f{}};
  int KT = K >> 5;

  // prologue: kt = 0 fragments
  v16bf a0 = load_frag2(Arow0, Arow0 + 8);
  v16bf a1 = load_frag2(Arow1, Arow1 + 8);
  v16bf b0 = load_frag2(Bbase, Bbase + 8);
  v16bf b1 = load_frag2(Bbase + 512, Bbase + 520);
  v16bf b2 = load_frag2(Bbase + 1024, Bbase + 1032);
  v16bf b3 = load_frag2(Bbase + 1536, Bbase + 1544);

#pragma unroll 2
  for (int kt = 0; kt < KT; ++kt) {
    int ktn = (kt + 1 < KT) ? (kt + 1) : kt;
    const unsigned short* An0 = Arow0 + ktn * 32;
    const unsigned short* An1 = Arow1 + ktn * 32;
    const unsigned short* Bn = Bbase + (size_t)ktn * bstep;
    __builtin_prefetch(Bbase + (size_t)(ktn + 4 < KT ? ktn + 4 : ktn) * bstep, 0, 1);
    // issue next-iteration loads before consuming current fragments
    v16bf na0 = load_frag2(An0, An0 + 8);
    v16bf na1 = load_frag2(An1, An1 + 8);
    v16bf nb0 = load_frag2(Bn, Bn + 8);
    v16bf nb1 = load_frag2(Bn + 512, Bn + 520);
    v16bf nb2 = load_frag2(Bn + 1024, Bn + 1032);
    v16bf nb3 = load_frag2(Bn + 1536, Bn + 1544);

    acc[0] = wmma_bf(a0, b0, acc[0]);
    acc[1] = wmma_bf(a0, b1, acc[1]);
    acc[2] = wmma_bf(a0, b2, acc[2]);
    acc[3] = wmma_bf(a0, b3, acc[3]);
    acc[4] = wmma_bf(a1, b0, acc[4]);
    acc[5] = wmma_bf(a1, b1, acc[5]);
    acc[6] = wmma_bf(a1, b2, acc[6]);
    acc[7] = wmma_bf(a1, b3, acc[7]);

    a0 = na0; a1 = na1; b0 = nb0; b1 = nb1; b2 = nb2; b3 = nb3;
  }

  int col = lane & 15;
#pragma unroll
  for (int mi = 0; mi < 2; ++mi) {
    int rbase = mt * 32 + mi * 16 + ((lane >> 4) << 3);
#pragma unroll
    for (int nt = 0; nt < 4; ++nt) {
      int gc = n0 + nt * 16 + col;
      float bv = bias ? bias[gc] : 0.f;
#pragma unroll
      for (int v = 0; v < 8; ++v) {
        int gr = rbase + v;
        float val = acc[mi * 4 + nt][v] + bv;
        if (act) val = 0.5f * val * (1.f + erff(val * 0.70710678118654752f));
        if (resid) val += resid[(size_t)gr * N + gc];
        if (outf) outf[(size_t)gr * N + gc] = val;
        if (outb) outb[(size_t)gr * N + gc] = f2bf(val);
      }
    }
  }
}

// ---------------- V transpose: Vb [bh][k][hd] -> Vt [bh][hd][k] (bf16) --------------
__global__ void __launch_bounds__(256, 1)
vtrans_kernel(const unsigned short* __restrict__ Vb, unsigned short* __restrict__ Vt) {
  size_t t = (size_t)blockIdx.x * 256 + threadIdx.x;   // t < B*H*HD*SKV
  int k = (int)(t & (SKV_ - 1));
  size_t rem = t >> 12;
  int hd = (int)(rem & (HD_ - 1));
  int bh = (int)(rem >> 6);
  Vt[t] = Vb[(size_t)bh * SKV_ * HD_ + (size_t)k * HD_ + hd];
}

// ---------------- S = scale * Q Kh^T + mask  ->  f32 A-region ----------------------
__global__ void __launch_bounds__(256, 1)
qk_kernel(const unsigned short* __restrict__ Qb,
          const unsigned short* __restrict__ Kb,
          const float* __restrict__ mask,
          float* __restrict__ Aout, float scale) {
  int lane = threadIdx.x & 31, w = threadIdx.x >> 5;
  const int perbh = (SQ_ / 16) * (SKV_ / 512);   // 256 blocks per (b,h)
  int bh = blockIdx.x / perbh;
  int r = blockIdx.x % perbh;
  int mt = r >> 3;
  int nc = r & 7;
  int b = bh >> 4;
  int n0 = (nc << 9) + (w << 6);
  const unsigned short* Qh = Qb + (size_t)bh * SQ_ * HD_;
  const unsigned short* Kh = Kb + (size_t)bh * SKV_ * HD_;
  float* Ao = Aout + (size_t)bh * SQ_ * SKV_;
  const float* mrow = mask + (size_t)b * SKV_;
  int m = lane & 15;
  int kbA = (lane >> 4) << 4;
  int kb2 = (lane >> 4) << 4;
  const unsigned short* Arow = Qh + (size_t)(mt * 16 + m) * HD_ + kbA;
  v16bf aq0 = load_frag2(Arow, Arow + 8);          // hd 0..31
  v16bf aq1 = load_frag2(Arow + 32, Arow + 40);    // hd 32..63
  int rb = mt * 16 + ((lane >> 4) << 3);
#pragma unroll
  for (int nt = 0; nt < 4; ++nt) {
    int colb = n0 + nt * 16;
    const unsigned short* Kp = Kh + (size_t)(colb + (lane & 15)) * HD_ + kb2;
    v16bf b0 = load_frag2(Kp, Kp + 8);
    v16bf b1 = load_frag2(Kp + 32, Kp + 40);
    v8f c = {};
    c = wmma_bf(aq0, b0, c);
    c = wmma_bf(aq1, b1, c);
    int gc = colb + (lane & 15);
    float mv = mrow[gc];
#pragma unroll
    for (int v = 0; v < 8; ++v)
      Ao[(size_t)(rb + v) * SKV_ + gc] = c[v] * scale + mv;
  }
}

// ---------------- row softmax in place over SKV (block per row) --------------------
__global__ void __launch_bounds__(256, 1)
softmax_kernel(float* __restrict__ Am) {
  float* r = Am + (size_t)blockIdx.x * SKV_;
  __shared__ float red[8];
  int tid = threadIdx.x, lane = tid & 31, w = tid >> 5;
  float mx = -3.402823466e38f;
  for (int i = tid; i < SKV_; i += 256) mx = fmaxf(mx, r[i]);
  for (int o = 16; o > 0; o >>= 1) mx = fmaxf(mx, __shfl_xor(mx, o, 32));
  if (lane == 0) red[w] = mx;
  __syncthreads();
  if (tid == 0) { float a = red[0]; for (int i = 1; i < 8; ++i) a = fmaxf(a, red[i]); red[0] = a; }
  __syncthreads();
  mx = red[0];
  __syncthreads();
  float s = 0.f;
  for (int i = tid; i < SKV_; i += 256) { float e = expf(r[i] - mx); r[i] = e; s += e; }
  for (int o = 16; o > 0; o >>= 1) s += __shfl_xor(s, o, 32);
  if (lane == 0) red[w] = s;
  __syncthreads();
  if (tid == 0) { float a = 0.f; for (int i = 0; i < 8; ++i) a += red[i]; red[0] = a; }
  __syncthreads();
  float inv = 1.f / red[0];
  for (int i = tid; i < SKV_; i += 256) r[i] *= inv;
}

// ------- O = A @ V -> bf16 attn output; register double-buffered K-loop -----------
__global__ void __launch_bounds__(256, 1)
av_kernel(const float* __restrict__ Am,
          const unsigned short* __restrict__ Vt,
          unsigned short* __restrict__ Ob) {
  int lane = threadIdx.x & 31, w = threadIdx.x >> 5;
  int gw = blockIdx.x * 8 + w;
  int bh = gw >> 5;                     // 32 m-tiles per (b,h)
  int mt = gw & 31;
  const float* Abh = Am + (size_t)bh * SQ_ * SKV_;
  const unsigned short* Vh = Vt + (size_t)bh * HD_ * SKV_;
  int m = lane & 15;
  int kbA = (lane >> 4) << 4;
  int kb2 = (lane >> 4) << 4;
  const float* Ar = Abh + (size_t)(mt * 16 + m) * SKV_ + kbA;
  const unsigned short* V0 = Vh + (size_t)(lane & 15) * SKV_ + kb2;
  const unsigned short* V1 = V0 + (size_t)16 * SKV_;
  const unsigned short* V2 = V0 + (size_t)32 * SKV_;
  const unsigned short* V3 = V0 + (size_t)48 * SKV_;

  v8f acc[4] = {v8f{}, v8f{}, v8f{}, v8f{}};
  const int KT = SKV_ >> 5;

  v16bf af = load_frag_f32(Ar);
  v16bf b0 = load_frag2(V0, V0 + 8);
  v16bf b1 = load_frag2(V1, V1 + 8);
  v16bf b2 = load_frag2(V2, V2 + 8);
  v16bf b3 = load_frag2(V3, V3 + 8);

#pragma unroll 2
  for (int kt = 0; kt < KT; ++kt) {
    int ktn = (kt + 1 < KT) ? (kt + 1) : kt;
    v16bf naf = load_frag_f32(Ar + ktn * 32);
    v16bf nb0 = load_frag2(V0 + ktn * 32, V0 + ktn * 32 + 8);
    v16bf nb1 = load_frag2(V1 + ktn * 32, V1 + ktn * 32 + 8);
    v16bf nb2 = load_frag2(V2 + ktn * 32, V2 + ktn * 32 + 8);
    v16bf nb3 = load_frag2(V3 + ktn * 32, V3 + ktn * 32 + 8);

    acc[0] = wmma_bf(af, b0, acc[0]);
    acc[1] = wmma_bf(af, b1, acc[1]);
    acc[2] = wmma_bf(af, b2, acc[2]);
    acc[3] = wmma_bf(af, b3, acc[3]);

    af = naf; b0 = nb0; b1 = nb1; b2 = nb2; b3 = nb3;
  }

  int rb = mt * 16 + ((lane >> 4) << 3);
  int col = lane & 15;
  unsigned short* Oh = Ob + (size_t)bh * SQ_ * HD_;
#pragma unroll
  for (int nt = 0; nt < 4; ++nt)
#pragma unroll
    for (int v = 0; v < 8; ++v)
      Oh[(size_t)(rb + v) * HD_ + nt * 16 + col] = f2bf(acc[nt][v]);
}

extern "C" void kernel_launch(void* const* d_in, const int* in_sizes, int n_in,
                              void* d_out, int out_size, void* d_ws, size_t ws_size,
                              hipStream_t stream) {
  const float* kv    = (const float*)d_in[0];
  const float* q     = (const float*)d_in[1];
  const float* mask  = (const float*)d_in[2];
  const float* lnkv_g= (const float*)d_in[3];
  const float* lnkv_b= (const float*)d_in[4];
  const float* lnq_g = (const float*)d_in[5];
  const float* lnq_b = (const float*)d_in[6];
  const float* Wq = (const float*)d_in[7];  const float* bq = (const float*)d_in[8];
  const float* Wk = (const float*)d_in[9];  const float* bk = (const float*)d_in[10];
  const float* Wv = (const float*)d_in[11]; const float* bv = (const float*)d_in[12];
  const float* Wo = (const float*)d_in[13]; const float* bo = (const float*)d_in[14];
  const float* lnf_g = (const float*)d_in[15];
  const float* lnf_b = (const float*)d_in[16];
  const float* W1 = (const float*)d_in[17]; const float* b1 = (const float*)d_in[18];
  const float* W2 = (const float*)d_in[19]; const float* b2 = (const float*)d_in[20];

  float* outF = (float*)d_out;                       // [B,SQ,D]
  float* Aout = outF + (size_t)B_ * SQ_ * D_;        // [B,H,SQ,SKV]

  char* ws = (char*)d_ws;
  size_t off = 0;
  auto alloc = [&](size_t bytes) -> void* {
    void* p = ws + off;
    off = (off + bytes + 255) & ~(size_t)255;
    return p;
  };
  const size_t MQ = (size_t)B_ * SQ_;     // 2048
  const size_t MKV = (size_t)B_ * SKV_;   // 16384
  unsigned short* qln  = (unsigned short*)alloc(MQ * D_ * 2);
  unsigned short* kvln = (unsigned short*)alloc(MKV * D_ * 2);
  unsigned short* Wqp  = (unsigned short*)alloc((size_t)D_ * D_ * 2);
  unsigned short* Wkp  = (unsigned short*)alloc((size_t)D_ * D_ * 2);
  unsigned short* Wvp  = (unsigned short*)alloc((size_t)D_ * D_ * 2);
  unsigned short* Wop  = (unsigned short*)alloc((size_t)D_ * D_ * 2);
  unsigned short* W1p  = (unsigned short*)alloc((size_t)D_ * FFW_ * 2);
  unsigned short* W2p  = (unsigned short*)alloc((size_t)FFW_ * D_ * 2);
  unsigned short* Qb   = (unsigned short*)alloc(MQ * D_ * 2);
  unsigned short* Kb   = (unsigned short*)alloc(MKV * D_ * 2);
  unsigned short* Vb   = (unsigned short*)alloc(MKV * D_ * 2);
  unsigned short* Vt   = (unsigned short*)alloc(MKV * D_ * 2);
  unsigned short* attnb= (unsigned short*)alloc(MQ * D_ * 2);
  float*          out1 = (float*)alloc(MQ * D_ * 4);
  unsigned short* hb   = (unsigned short*)alloc(MQ * D_ * 2);
  unsigned short* ffwb = (unsigned short*)alloc(MQ * FFW_ * 2);

  // 1) LayerNorms -> bf16
  ln_kernel<<<(int)MQ, 256, 0, stream>>>(q, lnq_g, lnq_b, qln, D_);
  ln_kernel<<<(int)MKV, 256, 0, stream>>>(kv, lnkv_g, lnkv_b, kvln, D_);

  // 2) Weight packing into WMMA B-fragment order
  packw_kernel<<<(D_ * D_) / 256, 256, 0, stream>>>(Wq, Wqp, D_);
  packw_kernel<<<(D_ * D_) / 256, 256, 0, stream>>>(Wk, Wkp, D_);
  packw_kernel<<<(D_ * D_) / 256, 256, 0, stream>>>(Wv, Wvp, D_);
  packw_kernel<<<(D_ * D_) / 256, 256, 0, stream>>>(Wo, Wop, D_);
  packw_kernel<<<(D_ * FFW_) / 256, 256, 0, stream>>>(W1, W1p, FFW_);
  packw_kernel<<<(FFW_ * D_) / 256, 256, 0, stream>>>(W2, W2p, D_);

  // 3) Q/K/V projections (bf16 WMMA, bf16 out); 32-row wave slabs
  gemm_bf16_kernel<<<(int)(MQ / 32) * (D_ / 512), 256, 0, stream>>>(
      qln, Wqp, bq, nullptr, nullptr, Qb, D_, D_, 0);
  gemm_bf16_kernel<<<(int)(MKV / 32) * (D_ / 512), 256, 0, stream>>>(
      kvln, Wkp, bk, nullptr, nullptr, Kb, D_, D_, 0);
  gemm_bf16_kernel<<<(int)(MKV / 32) * (D_ / 512), 256, 0, stream>>>(
      kvln, Wvp, bv, nullptr, nullptr, Vb, D_, D_, 0);

  // 4) V transpose per head: [bh][k][hd] -> [bh][hd][k]
  vtrans_kernel<<<(int)(MKV * D_ / 256), 256, 0, stream>>>(Vb, Vt);

  // 5) Attention scores S = scale*QK^T + mask  (written into d_out's A region)
  const float scale = 0.03125f;  // 1024^-0.5
  qk_kernel<<<B_ * H_ * (SQ_ / 16) * (SKV_ / 512), 256, 0, stream>>>(
      Qb, Kb, mask, Aout, scale);

  // 6) Softmax in place -> normalized A (this IS the second output)
  softmax_kernel<<<B_ * H_ * SQ_, 256, 0, stream>>>(Aout);

  // 7) O = A @ V -> bf16 per-head-contiguous == [B*SQ, D] row-major
  av_kernel<<<B_ * H_ * (SQ_ / 16) / 8, 256, 0, stream>>>(Aout, Vt, attnb);

  // 8) Wo projection + bias + residual(q) -> f32 out1
  gemm_bf16_kernel<<<(int)(MQ / 32) * (D_ / 512), 256, 0, stream>>>(
      attnb, Wop, bo, q, out1, nullptr, D_, D_, 0);

  // 9) Final layernorm -> bf16 h
  ln_kernel<<<(int)MQ, 256, 0, stream>>>(out1, lnf_g, lnf_b, hb, D_);

  // 10) MLP up + exact GELU -> bf16
  gemm_bf16_kernel<<<(int)(MQ / 32) * (FFW_ / 512), 256, 0, stream>>>(
      hb, W1p, b1, nullptr, nullptr, ffwb, FFW_, D_, 1);

  // 11) MLP down + bias + residual(out1) -> final f32 output
  gemm_bf16_kernel<<<(int)(MQ / 32) * (D_ / 512), 256, 0, stream>>>(
      ffwb, W2p, b2, out1, outF, nullptr, D_, FFW_, 0);
}
